// Llama4MoE_64244120814301
// MI455X (gfx1250) — matrix operand order repaired and checked by reading, not compiled, not run on hardware
//
#include <hip/hip_runtime.h>
#include <hip/hip_bf16.h>

// dims
#define TT  1024
#define HH  1024
#define II  2048
#define SSI 4096
#define EE  8

typedef __bf16 bf16t;
typedef __attribute__((ext_vector_type(16))) __bf16 v16bf;
typedef __attribute__((ext_vector_type(8)))  __bf16 v8bf;
typedef __attribute__((ext_vector_type(2)))  __bf16 v2bf;
typedef __attribute__((ext_vector_type(8)))  float  v8f;
typedef __attribute__((ext_vector_type(4)))  float  v4f;
typedef __attribute__((ext_vector_type(2)))  float  v2f;

// macro-tile
#define BM 128
#define BN 64
#define BK 32
#define KPAD 8   // LDS stride pad (halves); 40*2B = 80B keeps 16B alignment of 8-half chunks

// LDS byte offset of a shared-memory pointer (for async-to-LDS VDST operand)
__device__ __forceinline__ unsigned lds_addr(const void* p) {
  return (unsigned)(size_t)(__attribute__((address_space(3))) const void*)p;
}

// two f32 -> packed bf16x2; __builtin_convertvector lets the backend pick
// v_cvt_pk_bf16_f32 instead of scalar cvt + shift/or assembly
__device__ __forceinline__ unsigned pack_bf16x2(float lo, float hi) {
  v2f f;
  f[0] = lo;
  f[1] = hi;
  union { v2bf h; unsigned u; } p;
  p.h = __builtin_convertvector(f, v2bf);
  return p.u;
}

// ---------------------------------------------------------------------------
// Router: logits = x @ rw^T, top-1 argmax, w = sigmoid(best); build per-expert
// token lists with atomics (real dispatch instead of dense one-hot).
// ---------------------------------------------------------------------------
__global__ void moe_zero(int* counts) {
  if (threadIdx.x < EE) counts[threadIdx.x] = 0;
}

__global__ __launch_bounds__(256) void moe_router(
    const float* __restrict__ x, const float* __restrict__ rw,
    float* __restrict__ wt, int* __restrict__ counts, int* __restrict__ lists) {
  int t = blockIdx.x * blockDim.x + threadIdx.x;
  if (t >= TT) return;
  const float* xr = x + (size_t)t * HH;
  float best = -3.4e38f;
  int bi = 0;
  for (int e = 0; e < EE; ++e) {
    const float* w = rw + (size_t)e * HH;
    float acc = 0.f;
    for (int k = 0; k < HH; k += 4) {
      v4f xv = *(const v4f*)(xr + k);
      v4f wv = *(const v4f*)(w + k);
      acc += xv[0] * wv[0] + xv[1] * wv[1] + xv[2] * wv[2] + xv[3] * wv[3];
    }
    if (acc > best) { best = acc; bi = e; }  // first-max tie-break, like top_k
  }
  wt[t] = 1.f / (1.f + __expf(-best));
  int pos = atomicAdd(&counts[bi], 1);
  lists[bi * TT + pos] = t;
}

// x (f32) -> xb (bf16) for shared MLP, and route-weighted xw = w[t]*x (bf16).
// Packed v_cvt_pk_bf16_f32 + single 8B store per 4 elements per array.
__global__ __launch_bounds__(256) void moe_convert(
    const float* __restrict__ x, const float* __restrict__ wt,
    bf16t* __restrict__ xb, bf16t* __restrict__ xw) {
  int i = (blockIdx.x * 256 + threadIdx.x) * 4;
  if (i >= TT * HH) return;
  float w = wt[i / HH];
  v4f v = *(const v4f*)(x + i);
  union { unsigned long long u; unsigned w32[2]; } qa, qb;
  qa.w32[0] = pack_bf16x2(v[0], v[1]);
  qa.w32[1] = pack_bf16x2(v[2], v[3]);
  qb.w32[0] = pack_bf16x2(v[0] * w, v[1] * w);
  qb.w32[1] = pack_bf16x2(v[2] * w, v[3] * w);
  *(unsigned long long*)(xb + i) = qa.u;
  *(unsigned long long*)(xw + i) = qb.u;
}

// ---------------------------------------------------------------------------
// WMMA GEMM: C[M,N] = A[M,K](bf16) @ B[K,N](f32 weights -> bf16 in registers).
// 256 thr = 8 waves; block tile 128x64; each wave owns 32x32 (2x2 WMMA accs).
// A tile staged via global_load_async_to_lds_b128 (ASYNCcnt), double-buffered
// LDS, 2x-unrolled K loop (K/BK always even), one barrier per K-step.
// GATED fuses silu(g)*u; GATHER uses per-expert token lists.
// OUT_MODE: 0 = bf16 store (gated intermediate), 1 = f32 store, 2 = f32 +=.
// ---------------------------------------------------------------------------
template <bool GATED, bool GATHER, int OUT_MODE>
__global__ __launch_bounds__(256) void moe_wmma_gemm(
    const bf16t* __restrict__ A, const float* __restrict__ B0g,
    const float* __restrict__ B1g, void* __restrict__ OutV,
    const int* __restrict__ counts, const int* __restrict__ lists,
    int M, int N, int K, int ldA, int ldOut) {
  const int tid   = threadIdx.x;
  const int e     = GATHER ? blockIdx.z : 0;
  const int mTile = blockIdx.y;
  const int nTile = blockIdx.x;

  int cnt = M;
  if constexpr (GATHER) cnt = counts[e];
  if (mTile * BM >= cnt) return;  // empty expert tile: uniform early exit

  const float* __restrict__ B0 = B0g + (size_t)e * K * N;
  const float* __restrict__ B1 = nullptr;
  if constexpr (GATED) B1 = B1g + (size_t)e * K * N;

  __shared__ bf16t As [2][BM][BK + KPAD];
  __shared__ bf16t Bs0[2][BN][BK + KPAD];                    // stored [n][k]
  __shared__ bf16t Bs1[GATED ? 2 : 1][GATED ? BN : 1][BK + KPAD];
  __shared__ int   rowTab[BM];

  if (tid < BM) {
    int m = mTile * BM + tid;
    int r = -1;
    if (m < cnt) { if constexpr (GATHER) r = lists[e * TT + m]; else r = m; }
    rowTab[tid] = r;
  }
  __syncthreads();

  const int lane = tid & 31;
  const int wave = tid >> 5;
  const int wm   = wave & 3;   // 4 waves along M (4*32 = 128)
  const int wn   = wave >> 2;  // 2 waves along N (2*32 = 64)

  // ---- hoisted A staging state: 2 async b128 copies per thread per K-step
  const int arow0 = tid >> 2;        // rows r and r+64
  const int agrp  = (tid & 3) * 8;   // 8-half group within row
  const bf16t* aptr[2];
  bool     aval[2];
  unsigned aoff0[2], aoff1[2];
#pragma unroll
  for (int p = 0; p < 2; ++p) {
    int r   = arow0 + p * 64;
    int row = rowTab[r];
    aval[p] = row >= 0;
    aptr[p] = A + (size_t)(aval[p] ? row : 0) * ldA + agrp;
    aoff0[p] = lds_addr(&As[0][r][agrp]);
    aoff1[p] = lds_addr(&As[1][r][agrp]);
    if (!aval[p]) {                  // zero-fill padded gather rows once
      v8bf z = {};
      *(v8bf*)&As[0][r][agrp] = z;
      *(v8bf*)&As[1][r][agrp] = z;
    }
  }

  // ---- B staging: each thread covers 2 k-rows x 4 n-cols, packed b32 stores
  const int skr = (tid >> 4) * 2;    // k row pair: 0,2,..,30
  const int snc = (tid & 15) * 4;    // n col:      0,4,..,60
  const float* bp0 = B0 + (size_t)skr * N + nTile * BN + snc;
  const float* bp1 = nullptr;
  if constexpr (GATED) bp1 = B1 + (size_t)skr * N + nTile * BN + snc;

  v4f g0, g1, u0, u1;
  auto loadB = [&](int kk) {
    const float* p0 = bp0 + (size_t)kk * N;
    g0 = *(const v4f*)p0;
    g1 = *(const v4f*)(p0 + N);
    __builtin_prefetch(p0 + (size_t)BK * N, 0, 1);   // next tile into L2
    if constexpr (GATED) {
      const float* p1 = bp1 + (size_t)kk * N;
      u0 = *(const v4f*)p1;
      u1 = *(const v4f*)(p1 + N);
      __builtin_prefetch(p1 + (size_t)BK * N, 0, 1);
    }
  };
  auto storeB = [&](int b) {
#pragma unroll
    for (int j = 0; j < 4; ++j)
      *(unsigned*)&Bs0[b][snc + j][skr] = pack_bf16x2(g0[j], g1[j]);
    if constexpr (GATED) {
#pragma unroll
      for (int j = 0; j < 4; ++j)
        *(unsigned*)&Bs1[b][snc + j][skr] = pack_bf16x2(u0[j], u1[j]);
    }
  };
  auto issueA = [&](int kk, const unsigned* aoff) {
#pragma unroll
    for (int p = 0; p < 2; ++p)
      if (aval[p]) {
        unsigned long long gaddr = (unsigned long long)(aptr[p] + kk);
        asm volatile("global_load_async_to_lds_b128 %0, %1, off"
                     :: "v"(aoff[p]), "v"(gaddr)
                     : "memory");
      }
  };

  v8f accG[2][2] = {};
  v8f accU[2][2] = {};

  auto computeTile = [&](int b) {
    // fragments, contiguous k order (same for A and B):
    // A 16-bit 16x32: lanes 0-15 -> M=lane, K 0..15;         lanes 16-31 -> K 16..31
    // B 32x16:        lanes 0-15 -> N=lane, K 0..15;         lanes 16-31 -> K 16..31
    const int kh  = (lane >> 4) * 16;
    const int bkf = (lane >> 4) * 16;
    v16bf af[2], bfG[2], bfU[2];
#pragma unroll
    for (int i = 0; i < 2; ++i) {
      const bf16t* ap = &As[b][wm * 32 + i * 16 + (lane & 15)][0];
      v8bf alo = *(const v8bf*)(ap + kh);
      v8bf ahi = *(const v8bf*)(ap + kh + 8);
      af[i] = __builtin_shufflevector(alo, ahi, 0, 1, 2, 3, 4, 5, 6, 7, 8, 9,
                                      10, 11, 12, 13, 14, 15);
      const bf16t* bpg = &Bs0[b][wn * 32 + i * 16 + (lane & 15)][0];
      v8bf glo = *(const v8bf*)(bpg + bkf);
      v8bf ghi = *(const v8bf*)(bpg + bkf + 8);
      bfG[i] = __builtin_shufflevector(glo, ghi, 0, 1, 2, 3, 4, 5, 6, 7, 8, 9,
                                       10, 11, 12, 13, 14, 15);
      if constexpr (GATED) {
        const bf16t* bpu = &Bs1[b][wn * 32 + i * 16 + (lane & 15)][0];
        v8bf ulo = *(const v8bf*)(bpu + bkf);
        v8bf uhi = *(const v8bf*)(bpu + bkf + 8);
        bfU[i] = __builtin_shufflevector(ulo, uhi, 0, 1, 2, 3, 4, 5, 6, 7, 8, 9,
                                         10, 11, 12, 13, 14, 15);
      }
    }
#pragma unroll
    for (int at = 0; at < 2; ++at)
#pragma unroll
      for (int bt = 0; bt < 2; ++bt) {
        accG[at][bt] = __builtin_amdgcn_wmma_f32_16x16x32_bf16(
            false, af[at], false, bfG[bt], (short)0, accG[at][bt], false, false);
        if constexpr (GATED)
          accU[at][bt] = __builtin_amdgcn_wmma_f32_16x16x32_bf16(
              false, af[at], false, bfU[bt], (short)0, accU[at][bt], false, false);
      }
  };

  // ---- software-pipelined, double-buffered main loop (K/BK is even) ----
  issueA(0, aoff0);
  loadB(0);
  for (int kk = 0; kk < K; kk += 2 * BK) {
    // phase 0: buffer 0
    storeB(0);
    asm volatile("s_wait_asynccnt 0x0" ::: "memory");  // A[0] resident in LDS
    __syncthreads();
    issueA(kk + BK, aoff1);                            // kk+BK < K always
    loadB(kk + BK);
    computeTile(0);
    // phase 1: buffer 1
    storeB(1);
    asm volatile("s_wait_asynccnt 0x0" ::: "memory");  // A[1] resident in LDS
    __syncthreads();
    if (kk + 2 * BK < K) {
      issueA(kk + 2 * BK, aoff0);
      loadB(kk + 2 * BK);
    }
    computeTile(1);
  }

  // ---- epilogue (C/D layout: VGPR r -> M=r (lanes 0-15) / M=r+8 (16-31), N=lane%16)
#pragma unroll
  for (int at = 0; at < 2; ++at) {
#pragma unroll
    for (int bt = 0; bt < 2; ++bt) {
#pragma unroll
      for (int r = 0; r < 8; ++r) {
        int m   = wm * 32 + at * 16 + ((lane < 16) ? r : r + 8);
        int row = rowTab[m];
        if (row < 0) continue;
        int n = nTile * BN + wn * 32 + bt * 16 + (lane & 15);
        if constexpr (OUT_MODE == 0) {
          float g = accG[at][bt][r];
          float u = accU[at][bt][r];
          float s = g / (1.f + __expf(-g));  // silu
          ((bf16t*)OutV)[(size_t)row * ldOut + n] = (bf16t)(s * u);
        } else if constexpr (OUT_MODE == 1) {
          ((float*)OutV)[(size_t)row * ldOut + n] = accG[at][bt][r];
        } else {
          float* o = (float*)OutV + (size_t)row * ldOut + n;
          *o += accG[at][bt][r];  // token rows expert-disjoint (top-1): no atomics
        }
      }
    }
  }
}

// ---------------------------------------------------------------------------
extern "C" void kernel_launch(void* const* d_in, const int* in_sizes, int n_in,
                              void* d_out, int out_size, void* d_ws, size_t ws_size,
                              hipStream_t stream) {
  const float* x  = (const float*)d_in[0];  // [T,H]
  const float* rw = (const float*)d_in[1];  // [E,H]
  const float* sg = (const float*)d_in[2];  // [H,SI]
  const float* su = (const float*)d_in[3];  // [H,SI]
  const float* sd = (const float*)d_in[4];  // [SI,H]
  const float* eg = (const float*)d_in[5];  // [E,H,I]
  const float* eu = (const float*)d_in[6];  // [E,H,I]
  const float* ed = (const float*)d_in[7];  // [E,I,H]
  float* out = (float*)d_out;               // [T,H] f32

  // workspace layout (16B-aligned segments)
  char*  ws     = (char*)d_ws;
  int*   counts = (int*)ws;                     //   4 KB
  float* wt     = (float*)(ws + 4096);          //   4 KB
  int*   lists  = (int*)(ws + 8192);            // 128 KB  [E][T]
  bf16t* xb     = (bf16t*)(ws + 139264);        //   2 MB  x bf16
  bf16t* xw     = (bf16t*)(ws + 2236416);       //   2 MB  route-weighted x bf16
  bf16t* hs     = (bf16t*)(ws + 4333568);       //   8 MB  shared silu(g)*u [T,SI]
  bf16t* gb     = (bf16t*)(ws + 12722176);      //   4 MB  routed silu(g)*u [T,I]

  moe_zero<<<1, 32, 0, stream>>>(counts);
  moe_router<<<TT / 256, 256, 0, stream>>>(x, rw, wt, counts, lists);
  moe_convert<<<(TT * HH) / (256 * 4), 256, 0, stream>>>(x, wt, xb, xw);

  dim3 blk(256);
  // shared expert: gate/up fused, then down (f32 store into d_out)
  moe_wmma_gemm<true, false, 0><<<dim3(SSI / BN, TT / BM, 1), blk, 0, stream>>>(
      xb, sg, su, hs, nullptr, nullptr, TT, SSI, HH, HH, SSI);
  moe_wmma_gemm<false, false, 1><<<dim3(HH / BN, TT / BM, 1), blk, 0, stream>>>(
      hs, sd, nullptr, out, nullptr, nullptr, TT, HH, SSI, SSI, HH);
  // routed experts: gathered gate/up, then gathered down (+= into d_out)
  moe_wmma_gemm<true, true, 0><<<dim3(II / BN, TT / BM, EE), blk, 0, stream>>>(
      xw, eg, eu, gb, counts, lists, TT, II, HH, HH, II);
  moe_wmma_gemm<false, true, 2><<<dim3(HH / BN, TT / BM, EE), blk, 0, stream>>>(
      gb, ed, nullptr, out, counts, lists, TT, HH, II, II, HH);
}
